// Conv2d_11287174054575
// MI455X (gfx1250) — matrix operand order repaired and checked by reading, not compiled, not run on hardware
//
#include <hip/hip_runtime.h>
#include <hip/hip_bf16.h>
#include <math.h>

typedef __attribute__((ext_vector_type(16))) _Float16 v16h;
typedef __attribute__((ext_vector_type(8)))  float    v8f;

// Problem constants (fixed by the reference)
#define BATCH 8
#define CIN   32
#define HW    28
#define LPOS  784            // 28*28
#define COUT  64
#define CK    288            // 32*3*3
#define KCH   9              // 288 / 32 K-chunks per WMMA
#define NT    4              // 64 channels / 16
#define WS_A_HALVES (NT*KCH*32*16)   // 18432 halves = 36864 bytes

// ---------------------------------------------------------------------------
// Prep: swizzle fp32 weights into WMMA 16-bit A-matrix (16x32) VGPR layout
// (per N-tile, per K-chunk), and compute Sw[co] = sum_k w^2 in fp32.
// A layout (16-bit A 16x32): lane L (M = L%16, hi = L/16), element e:
//   k = kc*32 + hi*16 + e   (contiguous k order, matched by P's B layout)
// ---------------------------------------------------------------------------
__global__ void lp_prep_kernel(const float* __restrict__ w,
                               _Float16* __restrict__ wsA,
                               float* __restrict__ sw) {
    int id = blockIdx.x * blockDim.x + threadIdx.x;
    if (id < WS_A_HALVES) {
        int e    = id & 15;
        int lane = (id >> 4) & 31;
        int kc   = (id >> 9) % KCH;
        int nt   = id / (KCH * 512);
        int hi   = lane >> 4;
        int m    = lane & 15;
        int k    = kc * 32 + hi * 16 + e;
        int co   = nt * 16 + m;
        wsA[id]  = (_Float16)w[co * CK + k];
    } else if (id < WS_A_HALVES + COUT) {
        int co = id - WS_A_HALVES;
        float s = 0.f;
        for (int k = 0; k < CK; ++k) {
            float v = w[co * CK + k];
            s += v * v;
        }
        sw[co] = s;
    }
}

// ---------------------------------------------------------------------------
// Main fused kernel. Grid: 8 batches * 49 M-tiles = 392 blocks of 128 threads
// (4 waves). Shared f16 patch matrix P is built once, cooperatively, directly
// in the 32x16 16-bit WMMA B-layout:
//   P[(k>>4)*256 + m*16 + (k&15)]  == patch[m][k]
// so each wave's B fragment for chunk kc is a contiguous 32B/lane LDS read.
// Wave w computes D = A_w x B: A = 16 channels x K (preswizzled weights),
// B = K x 16 positions; 9 x v_wmma_f32_16x16x32_f16 accumulate K = 288.
// ---------------------------------------------------------------------------
__global__ void __launch_bounds__(128)
lp_conv_wmma_kernel(const float* __restrict__ x,
                    const _Float16* __restrict__ wsA,
                    const float* __restrict__ sw,
                    float* __restrict__ out) {
    __shared__ float tile[CIN][4][30];        // padded fp32 input rows h0-1..h0+2
    __shared__ __align__(32) _Float16 P[CK * 16]; // f16 patches in B-layout (9216 B)
    __shared__ float part[8][16];             // Sx partial sums
    __shared__ float sx[16];
    __shared__ float swl[COUT];

    const int blk = blockIdx.x;
    const int b   = blk / 49;
    const int mt  = blk % 49;
    const int l0  = mt * 16;
    const int h0  = l0 / HW;       // first output row in this tile
    const int ih0 = h0 - 1;        // global h of tile row 0

    const int tid    = threadIdx.x;
    const int lane   = tid & 31;
    const int wave   = tid >> 5;
    const int laneHi = lane >> 4;

    // ---- per-wave A fragments (pre-swizzled f16 weights), 32B/lane each ----
    v16h a[KCH];
    {
        const _Float16* wp = wsA + (size_t)(wave * KCH) * 512 + lane * 16;
#pragma unroll
        for (int kc = 0; kc < KCH; ++kc)
            a[kc] = *(const v16h*)(wp + kc * 512);
    }

    // ---- stage padded fp32 input tile (zero-filled halo), coalesced ----
    const float* xb = x + (size_t)b * CIN * LPOS;
    for (int s = tid; s < CIN * 4 * 30; s += 128) {
        int col = s % 30;              // iw = col - 1
        int row = (s / 30) & 3;
        int c   = s / 120;
        int ih  = ih0 + row;
        int iw  = col - 1;
        float v = 0.f;
        if (ih >= 0 && ih < HW && iw >= 0 && iw < HW)
            v = xb[c * LPOS + ih * HW + iw];
        tile[c][row][col] = v;
    }
    if (tid < COUT) swl[tid] = sw[tid];
    __syncthreads();

    // ---- cooperative build of P (f16, B-layout) + fused Sx partials ----
    {
        const int m  = tid & 15;            // position within tile
        const int g  = tid >> 4;            // k-stride group 0..7
        const int lm = l0 + m;
        const int hm = lm / HW, wm = lm % HW;
        const int r0 = hm - h0;             // 0 or 1
        const float* trow = &tile[0][r0][wm];

        float s = 0.f;
        int c = 0, r = g;                   // k = c*9 + r, starting at k = g
        for (int k = g; k < CK; k += 8) {
            int kh = r / 3;
            int kw = r - 3 * kh;
            float v = trow[c * 120 + kh * 30 + kw];
            s += v * v;
            P[((k >> 4) << 8) + (m << 4) + (k & 15)] = (_Float16)v;
            r += 8;
            if (r >= 9) { r -= 9; ++c; }
        }
        part[g][m] = s;
    }
    __syncthreads();

    // ---- Sx reduction (16 lanes) ----
    if (tid < 16) {
        float s = 0.f;
#pragma unroll
        for (int g = 0; g < 8; ++g) s += part[g][tid];
        sx[tid] = s;
    }

    // ---- WMMA accumulation over K = 288: contiguous B fragment loads ----
    v8f acc = {};
    {
        const _Float16* pb = P + lane * 16;   // (hi*16+m)*16 halves
#pragma unroll
        for (int kc = 0; kc < KCH; ++kc) {
            v16h bf = *(const v16h*)(pb + kc * 512);
            acc = __builtin_amdgcn_wmma_f32_16x16x32_f16(
                /*neg_a=*/false, a[kc], /*neg_b=*/false, bf,
                /*c_mod=*/(short)0, acc, /*reuse_a=*/false, /*reuse_b=*/false);
        }
    }
    __syncthreads();   // sx visible to all waves

    // ---- epilogue: y = -sqrt(Sx - 2*G + Sw), coalesced 64B store segments ----
    const int m = lane & 15;
    const int l = l0 + m;
    const float sxm = sx[m];
    float* ob = out + ((size_t)b * COUT) * LPOS + l;
#pragma unroll
    for (int r = 0; r < 8; ++r) {
        int co   = wave * 16 + laneHi * 8 + r;   // D row M = r + 8*laneHi
        float d2 = sxm + swl[co] - 2.f * acc[r];
        ob[(size_t)co * LPOS] = -sqrtf(fmaxf(d2, 0.f));
    }
}

// ---------------------------------------------------------------------------
extern "C" void kernel_launch(void* const* d_in, const int* in_sizes, int n_in,
                              void* d_out, int out_size, void* d_ws, size_t ws_size,
                              hipStream_t stream) {
    const float* x = (const float*)d_in[0];   // (8,32,28,28) fp32
    const float* w = (const float*)d_in[1];   // (64,32,3,3) fp32
    float* out = (float*)d_out;               // (8,64,28,28) fp32

    _Float16* wsA = (_Float16*)d_ws;                         // 36864 B
    float*    sw  = (float*)((char*)d_ws + WS_A_HALVES * 2); // 256 B

    int prep_threads = WS_A_HALVES + COUT;
    lp_prep_kernel<<<(prep_threads + 255) / 256, 256, 0, stream>>>(w, wsA, sw);
    lp_conv_wmma_kernel<<<BATCH * 49, 128, 0, stream>>>(x, wsA, sw, out);
}
